// MultiheadAttention_81793357185547
// MI455X (gfx1250) — matrix operand order repaired and checked by reading, not compiled, not run on hardware
//
#include <hip/hip_runtime.h>
#include <hip/hip_bf16.h>

typedef __bf16 bf16_t;
typedef __attribute__((ext_vector_type(16))) __bf16 v16bf;
typedef __attribute__((ext_vector_type(8)))  __bf16 v8bf;
typedef __attribute__((ext_vector_type(8)))  float  v8f;

#define BB   2
#define SS   2048
#define DIN  1024
#define DMODEL 1024
#define HH   16
#define DK   64
#define MTOT (BB*SS)          // 4096

// ---------------- WMMA helpers (CDNA5 gfx1250, wave32) ----------------

__device__ __forceinline__ v8f wmma_bf16(v16bf a, v16bf b, v8f c) {
    // 8 args: (neg_a, A, neg_b, B, c_mod, C, reuse_a, reuse_b)
    return __builtin_amdgcn_wmma_f32_16x16x32_bf16(false, a, false, b, (short)0, c, false, false);
}

// A fragment (16x32 bf16, MxK), row-major src, leading dim `ld`.
// lane half hl = lane>>4:
//   elems 0..7  = src[row][k0 + hl*8 + i]
//   elems 8..15 = src[row][k0 + 16 + hl*8 + i]
__device__ __forceinline__ v16bf load_a_frag(const bf16_t* base, int ld, int row, int k0, int hl) {
    const bf16_t* p = base + (size_t)row * ld + k0 + hl * 8;
    v8bf lo = *(const v8bf*)(p);
    v8bf hi = *(const v8bf*)(p + 16);
    v16bf r;
#pragma unroll
    for (int i = 0; i < 8; ++i) { r[i] = lo[i]; r[8 + i] = hi[i]; }
    return r;
}

// B fragment (32x16 bf16, KxN), B column-major == src row-major [N][K].
// lane: col = lane&15, elems 0..7 = src[col][k0 + hl*8 + i], elems 8..15 = src[col][k0 + 16 + hl*8 + i]
__device__ __forceinline__ v16bf load_b_frag(const bf16_t* base, int ld, int col, int k0, int hl) {
    const bf16_t* p = base + (size_t)col * ld + k0 + hl * 8;
    v8bf lo = *(const v8bf*)(p);
    v8bf hi = *(const v8bf*)(p + 16);
    v16bf r;
#pragma unroll
    for (int i = 0; i < 8; ++i) { r[i] = lo[i]; r[8 + i] = hi[i]; }
    return r;
}

__device__ __forceinline__ float redmax16(float v) {
    v = fmaxf(v, __shfl_xor(v, 1, 32));
    v = fmaxf(v, __shfl_xor(v, 2, 32));
    v = fmaxf(v, __shfl_xor(v, 4, 32));
    v = fmaxf(v, __shfl_xor(v, 8, 32));
    return v;
}
__device__ __forceinline__ float redsum16(float v) {
    v += __shfl_xor(v, 1, 32);
    v += __shfl_xor(v, 2, 32);
    v += __shfl_xor(v, 4, 32);
    v += __shfl_xor(v, 8, 32);
    return v;
}

// gfx1250 async copy global->LDS (per-lane 16B), tracked by ASYNCcnt.
// LDS byte address = low 32 bits of the generic shared pointer.
__device__ __forceinline__ void async_g2l_b128(void* lds_ptr, const void* gptr) {
    unsigned int loff = (unsigned int)(unsigned long long)lds_ptr;
    unsigned long long ga = (unsigned long long)gptr;
    asm volatile("global_load_async_to_lds_b128 %0, %1, off"
                 :: "v"(loff), "v"(ga) : "memory");
}
__device__ __forceinline__ void wait_asynccnt0() {
    asm volatile("s_wait_asynccnt 0x0" ::: "memory");
}

// ---------------- Stage 0: fp32 -> bf16 ----------------

__global__ void cvt_bf16_kernel(const float* __restrict__ src, bf16_t* __restrict__ dst, int n) {
    int i = blockIdx.x * blockDim.x + threadIdx.x;
    if (i < n) dst[i] = (bf16_t)src[i];
}

// ---------------- Stage 1: QKV GEMM (64x64 wave tiles) ----------------

__global__ __launch_bounds__(256) void qkv_gemm_kernel(
    const bf16_t* __restrict__ xb, const bf16_t* __restrict__ w,
    const float* __restrict__ bias,
    bf16_t* __restrict__ Q, bf16_t* __restrict__ K, bf16_t* __restrict__ Vt)
{
    const int lane = threadIdx.x & 31;
    const int wid  = threadIdx.x >> 5;
    const int wave = blockIdx.x * 8 + wid;
    const int NT   = (3 * DMODEL) / 64;      // 48 column tiles of 64
    const int n0   = (wave % NT) * 64;
    const int m0   = (wave / NT) * 64;
    const int n    = lane & 15, hl = lane >> 4;

    v8f acc[4][4];
#pragma unroll
    for (int i = 0; i < 4; ++i)
#pragma unroll
        for (int j = 0; j < 4; ++j) acc[i][j] = (v8f)0.0f;

    for (int k0 = 0; k0 < DIN; k0 += 32) {
        v16bf a[4];
#pragma unroll
        for (int ms = 0; ms < 4; ++ms)
            a[ms] = load_a_frag(xb, DIN, m0 + ms * 16 + n, k0, hl);
#pragma unroll
        for (int j = 0; j < 4; ++j) {
            v16bf bf = load_b_frag(w, DIN, n0 + j * 16 + n, k0, hl);
#pragma unroll
            for (int ms = 0; ms < 4; ++ms)
                acc[ms][j] = wmma_bf16(a[ms], bf, acc[ms][j]);
        }
    }

    const int seg  = n0 / 64;            // 0..47 ; one (head, q/k/v) segment
    const int head = seg / 3, sel = seg % 3;
#pragma unroll
    for (int ms = 0; ms < 4; ++ms)
#pragma unroll
        for (int j = 0; j < 4; ++j)
#pragma unroll
            for (int r = 0; r < 8; ++r) {
                int m = m0 + ms * 16 + r + 8 * hl;      // 0..4095
                int b = m >> 11, s = m & (SS - 1);
                int d = j * 16 + n;                      // 0..63
                float v = acc[ms][j][r] + bias[n0 + d];
                bf16_t bv = (bf16_t)v;
                size_t bh = (size_t)(b * HH + head);
                if (sel == 0)      Q[(bh * SS + s) * DK + d] = bv;
                else if (sel == 1) K[(bh * SS + s) * DK + d] = bv;
                else               Vt[(bh * DK + d) * SS + s] = bv;
            }
}

// ---------------- Stage 2: flash attention ----------------
// block = one (b,h) x 128 q rows (8 waves, one 16-row q tile each).
// K/V chunks (32 kv cols) staged to LDS with async copies, double-buffered.

__global__ __launch_bounds__(256) void attn_kernel(
    const bf16_t* __restrict__ Q, const bf16_t* __restrict__ K,
    const bf16_t* __restrict__ Vt, bf16_t* __restrict__ vals)
{
    __shared__ __align__(16) bf16_t kbuf[2][32 * 64];   // [kv row][d]
    __shared__ __align__(16) bf16_t vbuf[2][64 * 32];   // [d][kv row]
    __shared__ __align__(16) bf16_t pbuf[8][16 * 32];   // per-wave P staging

    const int tid  = threadIdx.x;
    const int lane = tid & 31;
    const int wid  = tid >> 5;
    const int nQB  = SS / 128;               // 16 q-blocks per (b,h)
    const int qb   = blockIdx.x % nQB;
    const int bh   = blockIdx.x / nQB;       // 0..31
    const int n = lane & 15, hl = lane >> 4;

    const bf16_t* q  = Q  + (size_t)bh * SS * DK;
    const bf16_t* k  = K  + (size_t)bh * SS * DK;
    const bf16_t* vt = Vt + (size_t)bh * DK * SS;
    bf16_t* pb = pbuf[wid];

    const int qrow = qb * 128 + wid * 16 + n;
    v16bf aq0 = load_a_frag(q, DK, qrow, 0,  hl);
    v16bf aq1 = load_a_frag(q, DK, qrow, 32, hl);

    v8f acc[4];
#pragma unroll
    for (int t = 0; t < 4; ++t) acc[t] = (v8f)0.0f;
    float mprev[8], lsum[8];
#pragma unroll
    for (int r = 0; r < 8; ++r) { mprev[r] = -1e30f; lsum[r] = 0.0f; }

    const int vr = tid >> 2, vp = tid & 3;   // V staging: row 0..63, 16B piece 0..3

    // prologue: stage chunk 0 into buffer 0
    async_g2l_b128(&kbuf[0][tid * 8], k + (size_t)0 * DK + tid * 8);
    async_g2l_b128(&vbuf[0][vr * 32 + vp * 8], vt + (size_t)vr * SS + 0 + vp * 8);

    const int NC = SS / 32;                  // 64 kv chunks
    for (int ci = 0; ci < NC; ++ci) {
        wait_asynccnt0();                    // own copies of chunk ci complete
        __syncthreads();                     // all waves' copies visible; prior reads retired
        if (ci + 1 < NC) {
            const int j0n = (ci + 1) * 32, nb = (ci + 1) & 1;
            async_g2l_b128(&kbuf[nb][tid * 8], k + (size_t)j0n * DK + tid * 8);
            async_g2l_b128(&vbuf[nb][vr * 32 + vp * 8], vt + (size_t)vr * SS + j0n + vp * 8);
        }
        const bf16_t* kb = kbuf[ci & 1];
        const bf16_t* vb = vbuf[ci & 1];

        v8f s0 = (v8f)0.0f, s1 = (v8f)0.0f;
        s0 = wmma_bf16(aq0, load_b_frag(kb, DK, n,      0,  hl), s0);
        s0 = wmma_bf16(aq1, load_b_frag(kb, DK, n,      32, hl), s0);
        s1 = wmma_bf16(aq0, load_b_frag(kb, DK, 16 + n, 0,  hl), s1);
        s1 = wmma_bf16(aq1, load_b_frag(kb, DK, 16 + n, 32, hl), s1);

        const float sc = 0.125f;             // 1/sqrt(64)
        float mnew[8], alpha[8];
#pragma unroll
        for (int r = 0; r < 8; ++r) {
            float a = s0[r] * sc, b = s1[r] * sc;
            s0[r] = a; s1[r] = b;
            float mx = redmax16(fmaxf(a, b));
            mnew[r]  = fmaxf(mprev[r], mx);
            alpha[r] = __expf(mprev[r] - mnew[r]);
        }
#pragma unroll
        for (int r = 0; r < 8; ++r) {
            float p0 = __expf(s0[r] - mnew[r]);
            float p1 = __expf(s1[r] - mnew[r]);
            s0[r] = p0; s1[r] = p1;
            float rs = redsum16(p0 + p1);
            lsum[r] = lsum[r] * alpha[r] + rs;
            mprev[r] = mnew[r];
#pragma unroll
            for (int t = 0; t < 4; ++t) acc[t][r] *= alpha[r];
        }

        // C layout -> per-wave LDS row-major [16 rows][32 kv cols]
#pragma unroll
        for (int r = 0; r < 8; ++r) {
            int row = r + 8 * hl;
            pb[row * 32 + n]      = (bf16_t)s0[r];
            pb[row * 32 + 16 + n] = (bf16_t)s1[r];
        }
        v16bf ap = load_a_frag(pb, 32, n, 0, hl);   // P as A fragment (16x32)
#pragma unroll
        for (int t = 0; t < 4; ++t) {
            v16bf bv = load_b_frag(vb, 32, t * 16 + n, 0, hl);
            acc[t] = wmma_bf16(ap, bv, acc[t]);
        }
        __syncthreads();                    // reads of this buffer done before rewrite
    }

    const int b = bh >> 4, head = bh & 15;
    float inv[8];
#pragma unroll
    for (int r = 0; r < 8; ++r) inv[r] = 1.0f / lsum[r];
#pragma unroll
    for (int t = 0; t < 4; ++t)
#pragma unroll
        for (int r = 0; r < 8; ++r) {
            int s   = qb * 128 + wid * 16 + r + 8 * hl;
            int col = head * 64 + t * 16 + n;
            vals[((size_t)(b * SS + s)) * DMODEL + col] = (bf16_t)(acc[t][r] * inv[r]);
        }
}

// ---------------- Stage 3: output projection (64x64 wave tiles) ----------------

__global__ __launch_bounds__(256) void oproj_kernel(
    const bf16_t* __restrict__ vals, const bf16_t* __restrict__ w,
    const float* __restrict__ bias, float* __restrict__ out)
{
    const int lane = threadIdx.x & 31;
    const int wid  = threadIdx.x >> 5;
    const int wave = blockIdx.x * 8 + wid;
    const int NT   = DMODEL / 64;            // 16
    const int n0   = (wave % NT) * 64;
    const int m0   = (wave / NT) * 64;
    const int n    = lane & 15, hl = lane >> 4;

    v8f acc[4][4];
#pragma unroll
    for (int i = 0; i < 4; ++i)
#pragma unroll
        for (int j = 0; j < 4; ++j) acc[i][j] = (v8f)0.0f;

    for (int k0 = 0; k0 < DMODEL; k0 += 32) {
        v16bf a[4];
#pragma unroll
        for (int ms = 0; ms < 4; ++ms)
            a[ms] = load_a_frag(vals, DMODEL, m0 + ms * 16 + n, k0, hl);
#pragma unroll
        for (int j = 0; j < 4; ++j) {
            v16bf bf = load_b_frag(w, DMODEL, n0 + j * 16 + n, k0, hl);
#pragma unroll
            for (int ms = 0; ms < 4; ++ms)
                acc[ms][j] = wmma_bf16(a[ms], bf, acc[ms][j]);
        }
    }
#pragma unroll
    for (int ms = 0; ms < 4; ++ms)
#pragma unroll
        for (int j = 0; j < 4; ++j)
#pragma unroll
            for (int r = 0; r < 8; ++r) {
                int m   = m0 + ms * 16 + r + 8 * hl;
                int col = n0 + j * 16 + n;
                out[(size_t)m * DMODEL + col] = acc[ms][j][r] + bias[col];
            }
}

// ---------------- launch ----------------

extern "C" void kernel_launch(void* const* d_in, const int* in_sizes, int n_in,
                              void* d_out, int out_size, void* d_ws, size_t ws_size,
                              hipStream_t stream) {
    const float* x     = (const float*)d_in[0];   // [2,2048,1024]
    const float* qkv_w = (const float*)d_in[1];   // [3072,1024]
    const float* qkv_b = (const float*)d_in[2];   // [3072]
    const float* o_w   = (const float*)d_in[3];   // [1024,1024]
    const float* o_b   = (const float*)d_in[4];   // [1024]
    float* out = (float*)d_out;

    char* ws = (char*)d_ws;
    bf16_t* xb   = (bf16_t*)(ws);                         // 8 MiB
    bf16_t* wqkv = (bf16_t*)(ws + (size_t)8388608);       // 6 MiB
    bf16_t* wo   = (bf16_t*)(ws + (size_t)14680064);      // 2 MiB
    bf16_t* Q    = (bf16_t*)(ws + (size_t)16777216);      // 8 MiB
    bf16_t* K    = (bf16_t*)(ws + (size_t)25165824);      // 8 MiB
    bf16_t* Vt   = (bf16_t*)(ws + (size_t)33554432);      // 8 MiB
    bf16_t* vals = (bf16_t*)(ws + (size_t)41943040);      // 8 MiB  (48 MiB total)

    const int NX = MTOT * DIN;           // 4194304
    const int NW = 3 * DMODEL * DIN;     // 3145728
    const int NO = DMODEL * DMODEL;      // 1048576
    cvt_bf16_kernel<<<(NX + 255) / 256, 256, 0, stream>>>(x, xb, NX);
    cvt_bf16_kernel<<<(NW + 255) / 256, 256, 0, stream>>>(qkv_w, wqkv, NW);
    cvt_bf16_kernel<<<(NO + 255) / 256, 256, 0, stream>>>(o_w, wo, NO);

    // 64 row-tiles * 48 col-tiles = 3072 waves / 8 per block
    qkv_gemm_kernel<<<384, 256, 0, stream>>>(xb, wqkv, qkv_b, Q, K, Vt);
    // 32 (b,h) * 16 q-blocks = 512 blocks of 8 waves
    attn_kernel<<<512, 256, 0, stream>>>(Q, K, Vt, vals);
    // 64 row-tiles * 16 col-tiles = 1024 waves / 8 per block
    oproj_kernel<<<128, 256, 0, stream>>>(vals, wo, o_b, out);
}
